// MAB_64587718197932
// MI455X (gfx1250) — matrix operand order repaired and checked by reading, not compiled, not run on hardware
//
#include <hip/hip_runtime.h>

typedef _Float16 half_t;
typedef __attribute__((ext_vector_type(16))) _Float16 v16h;
typedef __attribute__((ext_vector_type(8)))  float    v8f;
typedef __attribute__((ext_vector_type(4)))  float    f32x4;
typedef __attribute__((ext_vector_type(4)))  unsigned int u32x4;
typedef __attribute__((ext_vector_type(2)))  unsigned int u32x2;

union HFrag { v16h v; u32x4 q[2]; };
union H4 { half_t h[4]; u32x2 u; };
union H8 { half_t h[8]; u32x4 q; };

// A-fragment: halves 0..15 = row[klo .. klo+15] (contiguous K)
// (klo = (lane>>4)*16, matching the contiguous B fragment)
__device__ __forceinline__ v16h frag_rowK(const half_t* row, int klo) {
  HFrag f;
  f.q[0] = *(const u32x4*)(row + klo);
  f.q[1] = *(const u32x4*)(row + klo + 8);
  return f.v;
}
// B-fragment: 16 contiguous K values per lane (lanes 0-15: K=0..15, 16-31: K=16..31)
__device__ __forceinline__ v16h frag_contig(const half_t* p) {
  HFrag f;
  f.q[0] = *(const u32x4*)(p);
  f.q[1] = *(const u32x4*)(p + 8);
  return f.v;
}
__device__ __forceinline__ v8f wmma16(v16h a, v16h b, v8f c) {
  return __builtin_amdgcn_wmma_f32_16x16x32_f16(false, a, false, b, (short)0, c,
                                                false, false);
}
// CDNA5 async global->LDS copy (ASYNCcnt-tracked), per-lane LDS destinations.
__device__ __forceinline__ void async_load_b128(unsigned lds_off, const void* gaddr) {
  asm volatile("global_load_async_to_lds_b128 %0, %1, off"
               :: "v"(lds_off), "v"(gaddr)
               : "memory");
}
__device__ __forceinline__ void wait_async0() {
  asm volatile("s_wait_asynccnt 0" ::: "memory");
}

// 16-lane (DPP row) rotate reductions: pure VALU, no LDS crossbar traffic.
// row_ror:N ctrl = 0x120 + N; rotates within each 16-lane row independently.
template <int CTRL>
__device__ __forceinline__ float dpp_mov(float x) {
  return __int_as_float(
      __builtin_amdgcn_update_dpp(0, __float_as_int(x), CTRL, 0xf, 0xf, true));
}
__device__ __forceinline__ float rowmax16(float x) {
  x = fmaxf(x, dpp_mov<0x121>(x));
  x = fmaxf(x, dpp_mov<0x122>(x));
  x = fmaxf(x, dpp_mov<0x124>(x));
  x = fmaxf(x, dpp_mov<0x128>(x));
  return x;
}
__device__ __forceinline__ float rowsum16(float x) {
  x += dpp_mov<0x121>(x);
  x += dpp_mov<0x122>(x);
  x += dpp_mov<0x124>(x);
  x += dpp_mov<0x128>(x);
  return x;
}

// ---------------------------------------------------------------------------
// TN GEMM: out[m,n] = act( sum_k A[m,k] * W[n,k] + bias[n] ) (+ res[m,n])
// M = gridDim.y*128, N = 1024, K = 1024. fp32 in, f16 WMMA, f32 accumulate.
// Software-pipelined: global tile (kt+32) loads overlap WMMA on tile kt.
// ---------------------------------------------------------------------------
constexpr int LDA = 40;  // LDS row stride in halves (32 + 8 pad), rows 16B aligned

template <bool RELU, bool RES, bool ST32, bool ST16>
__global__ __launch_bounds__(256) void gemm_tn(
    const float* __restrict__ A, const float* __restrict__ W,
    const float* __restrict__ bias, const float* __restrict__ res,
    float* __restrict__ outf, half_t* __restrict__ outh, int K, float hscale) {
  __shared__ half_t As[128 * LDA];
  __shared__ half_t Bs[128 * LDA];
  const int t = threadIdx.x;
  const int lane = t & 31, wave = t >> 5;
  const int hi = lane >> 4, ln = lane & 15;
  const int waveM = wave >> 1, waveN = wave & 1;
  const long m0 = (long)blockIdx.y * 128;
  const int  n0 = blockIdx.x * 128;
  const int sr = t >> 1;         // staged row 0..127 (2 threads per row)
  const int sc = (t & 1) * 16;   // staged col base {0,16}

  const float* ap = A + (m0 + sr) * (long)K + sc;
  const float* wp = W + (n0 + sr) * (long)K + sc;

  v8f acc[2][4] = {};
  f32x4 aR[4], wR[4];
#pragma unroll
  for (int i = 0; i < 4; ++i) {
    aR[i] = *(const f32x4*)(ap + 4 * i);
    wR[i] = *(const f32x4*)(wp + 4 * i);
  }

  for (int kt = 0; kt < K; kt += 32) {
#pragma unroll
    for (int i = 0; i < 4; ++i) {
      H4 ah, wh;
      ah.h[0] = (half_t)aR[i].x; ah.h[1] = (half_t)aR[i].y;
      ah.h[2] = (half_t)aR[i].z; ah.h[3] = (half_t)aR[i].w;
      wh.h[0] = (half_t)wR[i].x; wh.h[1] = (half_t)wR[i].y;
      wh.h[2] = (half_t)wR[i].z; wh.h[3] = (half_t)wR[i].w;
      *(u32x2*)&As[sr * LDA + sc + 4 * i] = ah.u;
      *(u32x2*)&Bs[sr * LDA + sc + 4 * i] = wh.u;
    }
    __syncthreads();

    if (kt + 32 < K) {  // pipeline next tile behind the WMMA phase
#pragma unroll
      for (int i = 0; i < 4; ++i) {
        aR[i] = *(const f32x4*)(ap + kt + 32 + 4 * i);
        wR[i] = *(const f32x4*)(wp + kt + 32 + 4 * i);
      }
      __builtin_prefetch(ap + kt + 64, 0, 1);
      __builtin_prefetch(wp + kt + 64, 0, 1);
    }

    v16h af[2];
#pragma unroll
    for (int mt = 0; mt < 2; ++mt)
      af[mt] = frag_rowK(&As[(waveM * 32 + mt * 16 + ln) * LDA], hi * 16);
#pragma unroll
    for (int nt = 0; nt < 4; ++nt) {
      v16h bf = frag_contig(&Bs[(waveN * 64 + nt * 16 + ln) * LDA + hi * 16]);
#pragma unroll
      for (int mt = 0; mt < 2; ++mt) acc[mt][nt] = wmma16(af[mt], bf, acc[mt][nt]);
    }
    __syncthreads();
  }

#pragma unroll
  for (int mt = 0; mt < 2; ++mt)
#pragma unroll
    for (int nt = 0; nt < 4; ++nt) {
      const int gn = n0 + waveN * 64 + nt * 16 + ln;
      const float bs = bias[gn];
#pragma unroll
      for (int r = 0; r < 8; ++r) {
        const long gm = m0 + waveM * 32 + mt * 16 + r + 8 * hi;
        float v = acc[mt][nt][r] + bs;
        if (RELU) v = fmaxf(v, 0.0f);
        if (RES)  v += res[gm * 1024 + gn];
        if (ST32) outf[gm * 1024 + gn] = v;
        if (ST16) outh[gm * 1024 + gn] = (half_t)(v * hscale);
      }
    }
}

// ---------------------------------------------------------------------------
// Flash attention: grid = (Sq/128, H, B), block = 256 (8 waves x 16 q-rows).
// dh = 64, 32 KV rows/iter. Q was pre-scaled by (1/sqrt(1024))*log2(e), so the
// online softmax runs in the exp2 domain. K tiles arrive by async global->LDS
// (double-buffered, one barrier per iteration); V is transposed via registers.
// Row reductions use DPP row_ror rotations (groups are exactly 16-lane rows).
// ---------------------------------------------------------------------------
constexpr int KSS = 72;  // K-tile LDS stride (64 + 8)
constexpr int VTS = 40;  // V^T LDS stride   (32 + 8)
constexpr int PSS = 40;  // P-tile LDS stride (32 + 8)

__global__ __launch_bounds__(256) void flash_attn(
    const half_t* __restrict__ Qh, const half_t* __restrict__ Kh,
    const half_t* __restrict__ Vh, float* __restrict__ attn) {
  __shared__ half_t Ks[2][32 * KSS];
  __shared__ half_t VTs[2][64 * VTS];
  __shared__ half_t Ps[8 * 16 * PSS];
  const int t = threadIdx.x;
  const int lane = t & 31, wave = t >> 5;
  const int hi = lane >> 4, ln = lane & 15;
  const int hh = blockIdx.y, b = blockIdx.z;
  const long rowQ = (long)b * 1024 + blockIdx.x * 128 + wave * 16;
  const int dcol = hh * 64;
  const int sr = t >> 3, seg = t & 7;   // staging: row 0..31, 16B segment 0..7
  const long kvRow0 = (long)b * 1024;

  // resident q fragments: 16 rows x dh=64 as two K=32 fragments (pre-scaled)
  v16h qf[2];
  {
    const half_t* qrow = Qh + (rowQ + ln) * 1024 + dcol;
    qf[0] = frag_rowK(qrow, hi * 16);
    qf[1] = frag_rowK(qrow + 32, hi * 16);
  }

  v8f O[4] = {};
  float mi[8], li[8];
#pragma unroll
  for (int r = 0; r < 8; ++r) { mi[r] = -1e30f; li[r] = 0.0f; }

  // prologue: async K tile 0, V regs for tile 0
  async_load_b128((unsigned)(size_t)&Ks[0][sr * KSS + seg * 8],
                  Kh + (kvRow0 + sr) * 1024 + dcol + seg * 8);
  H8 vreg;
  vreg.q = *(const u32x4*)(Vh + (kvRow0 + sr) * 1024 + dcol + seg * 8);

  for (int kv = 0; kv < 1024; kv += 32) {
    const int buf = (kv >> 5) & 1;
    // scatter V rows into transposed LDS tile
#pragma unroll
    for (int i = 0; i < 8; ++i) VTs[buf][(seg * 8 + i) * VTS + sr] = vreg.h[i];
    wait_async0();           // this wave's K tile landed in LDS
    __syncthreads();         // all waves' tiles visible; prev buffers free

    if (kv + 32 < 1024) {    // prefetch next KV block behind the WMMA phase
      const long gr = kvRow0 + kv + 32 + sr;
      async_load_b128((unsigned)(size_t)&Ks[buf ^ 1][sr * KSS + seg * 8],
                      Kh + gr * 1024 + dcol + seg * 8);
      vreg.q = *(const u32x4*)(Vh + gr * 1024 + dcol + seg * 8);
    }

    // GEMM1: S(16x32) = q(16x64) . k^T   (already in log2 domain)
    v8f s0 = {}, s1 = {};
#pragma unroll
    for (int ks = 0; ks < 2; ++ks) {
      v16h b0 = frag_contig(&Ks[buf][ln * KSS + ks * 32 + hi * 16]);
      s0 = wmma16(qf[ks], b0, s0);
      v16h b1 = frag_contig(&Ks[buf][(16 + ln) * KSS + ks * 32 + hi * 16]);
      s1 = wmma16(qf[ks], b1, s1);
    }

    // online softmax per row (row m = r + 8*hi lives in one 16-lane DPP row)
    half_t* prow0 = &Ps[(wave * 16) * PSS];
#pragma unroll
    for (int r = 0; r < 8; ++r) {
      const float v0 = s0[r], v1 = s1[r];
      const float mx = rowmax16(fmaxf(v0, v1));
      const float mnew = fmaxf(mi[r], mx);
      const float corr = exp2f(mi[r] - mnew);
      const float p0 = exp2f(v0 - mnew);
      const float p1 = exp2f(v1 - mnew);
      const float rs = rowsum16(p0 + p1);
      li[r] = li[r] * corr + rs;
      mi[r] = mnew;
#pragma unroll
      for (int dt = 0; dt < 4; ++dt) O[dt][r] *= corr;
      const int m = r + 8 * hi;
      prow0[m * PSS + ln]      = (half_t)p0;
      prow0[m * PSS + 16 + ln] = (half_t)p1;
    }

    // GEMM2: O(16x64) += P(16x32) . V(32x64)
    v16h pf = frag_rowK(&prow0[ln * PSS], hi * 16);
#pragma unroll
    for (int dt = 0; dt < 4; ++dt) {
      v16h vf = frag_contig(&VTs[buf][(dt * 16 + ln) * VTS + hi * 16]);
      O[dt] = wmma16(pf, vf, O[dt]);
    }
  }

#pragma unroll
  for (int r = 0; r < 8; ++r) {
    const float inv = 1.0f / li[r];
    const long gm = rowQ + r + 8 * hi;
#pragma unroll
    for (int dt = 0; dt < 4; ++dt)
      attn[gm * 1024 + dcol + dt * 16 + ln] = O[dt][r] * inv;
  }
}

// ---------------------------------------------------------------------------
// Row LayerNorm over 1024 features, optional fused residual add (add != null)
// ---------------------------------------------------------------------------
__global__ __launch_bounds__(256) void lnorm(
    const float* __restrict__ x, const float* __restrict__ add,
    const float* __restrict__ g, const float* __restrict__ bt,
    float* __restrict__ out) {
  const long row = blockIdx.x;
  const int t = threadIdx.x;
  const int lane = t & 31, wave = t >> 5;
  __shared__ float sh1[8], sh2[8];
  float v[4];
  float s = 0.0f, s2 = 0.0f;
#pragma unroll
  for (int i = 0; i < 4; ++i) {
    const int c = t + i * 256;
    float a = x[row * 1024 + c];
    if (add) a += add[row * 1024 + c];
    v[i] = a; s += a; s2 += a * a;
  }
#pragma unroll
  for (int off = 16; off; off >>= 1) {
    s  += __shfl_xor(s, off, 32);
    s2 += __shfl_xor(s2, off, 32);
  }
  if (lane == 0) { sh1[wave] = s; sh2[wave] = s2; }
  __syncthreads();
  float ts = 0.0f, ts2 = 0.0f;
#pragma unroll
  for (int i = 0; i < 8; ++i) { ts += sh1[i]; ts2 += sh2[i]; }
  const float mu  = ts * (1.0f / 1024.0f);
  const float var = ts2 * (1.0f / 1024.0f) - mu * mu;
  const float inv = rsqrtf(var + 1e-5f);
#pragma unroll
  for (int i = 0; i < 4; ++i) {
    const int c = t + i * 256;
    out[row * 1024 + c] = (v[i] - mu) * inv * g[c] + bt[c];
  }
}

// ---------------------------------------------------------------------------
extern "C" void kernel_launch(void* const* d_in, const int* in_sizes, int n_in,
                              void* d_out, int out_size, void* d_ws,
                              size_t ws_size, hipStream_t stream) {
  (void)in_sizes; (void)n_in; (void)out_size; (void)ws_size;
  const float* Q  = (const float*)d_in[0];
  const float* Vv = (const float*)d_in[1];
  const float* Wq = (const float*)d_in[2];
  const float* bq = (const float*)d_in[3];
  const float* Wk = (const float*)d_in[4];
  const float* bk = (const float*)d_in[5];
  const float* Wv = (const float*)d_in[6];
  const float* bv = (const float*)d_in[7];
  const float* Wo = (const float*)d_in[8];
  const float* bo = (const float*)d_in[9];
  const float* g1 = (const float*)d_in[10];
  const float* b1 = (const float*)d_in[11];
  const float* g2 = (const float*)d_in[12];
  const float* b2 = (const float*)d_in[13];

  const size_t MN = (size_t)8192 * 1024;
  char* w = (char*)d_ws;
  float*  Qp   = (float*)w;  w += MN * 4;
  float*  attn = (float*)w;  w += MN * 4;
  float*  O1   = (float*)w;  w += MN * 4;
  float*  O2   = (float*)w;  w += MN * 4;
  half_t* Qh   = (half_t*)w; w += MN * 2;
  half_t* Kh   = (half_t*)w; w += MN * 2;
  half_t* Vh   = (half_t*)w; w += MN * 2;

  // (1/sqrt(DIM_V)) * log2(e): runs softmax in the exp2 domain for free
  const float qs = 0.03125f * 1.44269504088896f;

  dim3 gg(8, 64), blk(256);
  gemm_tn<false, false, true,  true ><<<gg, blk, 0, stream>>>(Q,  Wq, bq, nullptr, Qp,      Qh, 1024, qs);
  gemm_tn<false, false, false, true ><<<gg, blk, 0, stream>>>(Vv, Wk, bk, nullptr, nullptr, Kh, 1024, 1.0f);
  gemm_tn<false, false, false, true ><<<gg, blk, 0, stream>>>(Vv, Wv, bv, nullptr, nullptr, Vh, 1024, 1.0f);
  flash_attn<<<dim3(8, 16, 8), blk, 0, stream>>>(Qh, Kh, Vh, attn);
  lnorm<<<dim3(8192), blk, 0, stream>>>(Qp, attn, g1, b1, O1);
  gemm_tn<true, true, true, false><<<gg, blk, 0, stream>>>(O1, Wo, bo, O1, O2, nullptr, 1024, 1.0f);
  lnorm<<<dim3(8192), blk, 0, stream>>>(O2, nullptr, g2, b2, (float*)d_out);
}
